// CausalWaveletFieldAttentionI_82729660056310
// MI455X (gfx1250) — compile-verified, matrix operand order checked by reading, not run on hardware
//
#include <hip/hip_runtime.h>
#include <cstdint>
#include <cstddef>

// Problem constants (match reference)
#define Bq 4
#define Nq 2048
#define Dq 1024
#define Hq 16
#define HDq 64
#define Sq 11
#define Mq (Bq * Nq)   // 8192 rows for all GEMMs
#define Kq Dq          // 1024 reduction dim

typedef __attribute__((ext_vector_type(16))) __bf16 v16bf;
typedef __attribute__((ext_vector_type(8)))  __bf16 v8bf;
typedef __attribute__((ext_vector_type(8)))  float  v8f;
typedef __attribute__((ext_vector_type(4)))  int    v4i;

// Probe for gfx1250 async global->LDS path (ASYNCcnt-tracked).
#if defined(__has_builtin)
#if __has_builtin(__builtin_amdgcn_global_load_async_to_lds_b128) && \
    __has_builtin(__builtin_amdgcn_s_wait_asynccnt)
#define HAVE_ASYNC_LDS 1
#endif
#endif
#ifndef HAVE_ASYNC_LDS
#define HAVE_ASYNC_LDS 0
#endif

// Builtin wants typed v4i pointers: global (AS1) source, LDS (AS3) dest.
#define AS1V4(p) ((__attribute__((address_space(1))) v4i*)(v4i*)(size_t)(p))
#define AS3V4(p) ((__attribute__((address_space(3))) v4i*)(v4i*)(p))

// ---------------------------------------------------------------------------
// fp32 -> bf16 conversion (all sizes here are multiples of 4)
// ---------------------------------------------------------------------------
__global__ void cvt_bf16_kernel(const float* __restrict__ in,
                                __bf16* __restrict__ out, int n) {
  int i = (blockIdx.x * blockDim.x + threadIdx.x) * 4;
  if (i + 3 < n) {
    float4 v = *(const float4*)(in + i);
    out[i + 0] = (__bf16)v.x;
    out[i + 1] = (__bf16)v.y;
    out[i + 2] = (__bf16)v.z;
    out[i + 3] = (__bf16)v.w;
  } else {
    for (int k = 0; k < 4 && i + k < n; ++k) out[i + k] = (__bf16)in[i + k];
  }
}

// ---------------------------------------------------------------------------
// bf16 WMMA GEMM: C[M,Nc] = A[M,K] * W[Nc,K]^T + bias, optional sigmoid.
// Block tile 128(M) x 64(N), BK=64 (two v_wmma K-steps per stage).
// 8 waves: 4 along M x 2 along N, each wave owns a 32x32 quadrant (4 accs).
// Double-buffered LDS fed by GLOBAL_LOAD_ASYNC_TO_LDS_B128 when available.
// ---------------------------------------------------------------------------
#define BM 128
#define BN 64
#define BK 64
#define LDK 72  // +8 bf16 pad: rows stay 16B-aligned, banks de-conflicted

__launch_bounds__(256)
__global__ void gemm_bf16_wmma_kernel(const __bf16* __restrict__ A,   // [M,K]
                                      const __bf16* __restrict__ W,   // [Nc,K]
                                      const float* __restrict__ bias, // [Nc]
                                      float* __restrict__ C,          // [M,Nc]
                                      int M, int Nc, int K, int act) {
  __shared__ __bf16 As[2][BM * LDK];
  __shared__ __bf16 Bs[2][BN * LDK];

  const int t      = threadIdx.x;
  const int lane   = t & 31;
  const int wave   = t >> 5;
  const int wm     = (wave & 3) * 32;   // wave's M offset inside tile
  const int wn     = (wave >> 2) * 32;  // wave's N offset inside tile
  const int blockM = blockIdx.x * BM;
  const int blockN = blockIdx.y * BN;
  const int l16    = lane & 15;
  const int lh     = lane >> 4;

  v8f acc[2][2] = {};

#if HAVE_ASYNC_LDS
  // --- async double-buffered pipeline -------------------------------------
  auto issue = [&](int k0, __bf16* Abuf, __bf16* Bbuf) {
#pragma unroll
    for (int i = 0; i < 4; ++i) {                // A tile: 128x64 = 1024 chunks
      int chunk = t + i * 256;
      int row = chunk >> 3, cg = chunk & 7;
      __builtin_amdgcn_global_load_async_to_lds_b128(
          AS1V4(A + (size_t)(blockM + row) * K + k0 + cg * 8),
          AS3V4(&Abuf[row * LDK + cg * 8]), 0, 0);
    }
#pragma unroll
    for (int i = 0; i < 2; ++i) {                // B tile: 64x64 = 512 chunks
      int chunk = t + i * 256;
      int row = chunk >> 3, cg = chunk & 7;
      __builtin_amdgcn_global_load_async_to_lds_b128(
          AS1V4(W + (size_t)(blockN + row) * K + k0 + cg * 8),
          AS3V4(&Bbuf[row * LDK + cg * 8]), 0, 0);
    }
  };

  const int T = K / BK;
  issue(0, As[0], Bs[0]);
  if (T > 1) issue(BK, As[1], Bs[1]);

  for (int it = 0; it < T; ++it) {
    // 6 async ops per tile per thread; loads retire in order.
    if (it + 1 < T) __builtin_amdgcn_s_wait_asynccnt(6);
    else            __builtin_amdgcn_s_wait_asynccnt(0);
    __syncthreads();

    const __bf16* Ab = As[it & 1];
    const __bf16* Bb = Bs[it & 1];
#pragma unroll
    for (int kk = 0; kk < 2; ++kk) {
      v16bf af[2], bfr[2];
#pragma unroll
      for (int r = 0; r < 2; ++r)
        af[r] = *(const v16bf*)(&Ab[(wm + r * 16 + l16) * LDK + kk * 32 + lh * 16]);
#pragma unroll
      for (int c = 0; c < 2; ++c)
        bfr[c] = *(const v16bf*)(&Bb[(wn + c * 16 + l16) * LDK + kk * 32 + lh * 16]);
#pragma unroll
      for (int r = 0; r < 2; ++r)
#pragma unroll
        for (int c = 0; c < 2; ++c)
          acc[r][c] = __builtin_amdgcn_wmma_f32_16x16x32_bf16(
              false, af[r], false, bfr[c], (short)0, acc[r][c], false, false);
    }
    __syncthreads();
    if (it + 2 < T) issue((it + 2) * BK, As[it & 1], Bs[it & 1]);
  }
#else
  // --- fallback: batched loads into temps, single LDS stage ---------------
  for (int k0 = 0; k0 < K; k0 += BK) {
    v8bf tA[4], tB[2];
#pragma unroll
    for (int i = 0; i < 4; ++i) {
      int chunk = t + i * 256;
      int row = chunk >> 3, cg = chunk & 7;
      tA[i] = *(const v8bf*)(A + (size_t)(blockM + row) * K + k0 + cg * 8);
    }
#pragma unroll
    for (int i = 0; i < 2; ++i) {
      int chunk = t + i * 256;
      int row = chunk >> 3, cg = chunk & 7;
      tB[i] = *(const v8bf*)(W + (size_t)(blockN + row) * K + k0 + cg * 8);
    }
    if (k0 + BK < K)
      __builtin_prefetch(A + (size_t)(blockM + (t >> 3)) * K + k0 + BK, 0, 1);
    __syncthreads();  // previous stage fully consumed
#pragma unroll
    for (int i = 0; i < 4; ++i) {
      int chunk = t + i * 256;
      int row = chunk >> 3, cg = chunk & 7;
      *(v8bf*)(&As[0][row * LDK + cg * 8]) = tA[i];
    }
#pragma unroll
    for (int i = 0; i < 2; ++i) {
      int chunk = t + i * 256;
      int row = chunk >> 3, cg = chunk & 7;
      *(v8bf*)(&Bs[0][row * LDK + cg * 8]) = tB[i];
    }
    __syncthreads();
#pragma unroll
    for (int kk = 0; kk < 2; ++kk) {
      v16bf af[2], bfr[2];
#pragma unroll
      for (int r = 0; r < 2; ++r)
        af[r] = *(const v16bf*)(&As[0][(wm + r * 16 + l16) * LDK + kk * 32 + lh * 16]);
#pragma unroll
      for (int c = 0; c < 2; ++c)
        bfr[c] = *(const v16bf*)(&Bs[0][(wn + c * 16 + l16) * LDK + kk * 32 + lh * 16]);
#pragma unroll
      for (int r = 0; r < 2; ++r)
#pragma unroll
        for (int c = 0; c < 2; ++c)
          acc[r][c] = __builtin_amdgcn_wmma_f32_16x16x32_bf16(
              false, af[r], false, bfr[c], (short)0, acc[r][c], false, false);
    }
  }
#endif

  // Epilogue: C/D layout is VGPR i, lane l -> M = i + 8*(l>>4), N = l&15.
#pragma unroll
  for (int r = 0; r < 2; ++r) {
#pragma unroll
    for (int c = 0; c < 2; ++c) {
      int gn     = blockN + wn + c * 16 + l16;
      int gmBase = blockM + wm + r * 16 + lh * 8;
      float bv   = bias[gn];
#pragma unroll
      for (int i = 0; i < 8; ++i) {
        float v = acc[r][c][i] + bv;
        if (act == 1) v = 1.0f / (1.0f + __expf(-v));
        C[(size_t)(gmBase + i) * Nc + gn] = v;
      }
    }
  }
}

// ---------------------------------------------------------------------------
// Per-position: gains softmax (q·Wqs^T + prior), k-magnitude, field deposit.
// One wave32 per (b,h,n); lane covers 2 of the 64 head-dim elements.
// ---------------------------------------------------------------------------
__launch_bounds__(256)
__global__ void gains_field_kernel(const float* __restrict__ qkv,        // [B*N, 3D]
                                   const float* __restrict__ scale_gain, // [S,H]
                                   const float* __restrict__ Wqs,        // [S,HD]
                                   float* __restrict__ gains,            // [B,H,N,S]
                                   float* __restrict__ field) {          // [B,H,N,HD]
  int wave = threadIdx.x >> 5;
  int lane = threadIdx.x & 31;
  long pos = (long)blockIdx.x * 8 + wave;  // enumerates (b,h,n)
  int n    = (int)(pos % Nq);
  long bh  = pos / Nq;
  int h    = (int)(bh % Hq);
  int b    = (int)(bh / Hq);

  size_t rowbase = ((size_t)(b * Nq + n)) * (3 * Dq) + h * HDq;
  const float* qp = qkv + rowbase;
  const float* kp = qp + Dq;
  const float* vp = qp + 2 * Dq;

  float k0 = kp[lane], k1 = kp[lane + 32];
  float ss = k0 * k0 + k1 * k1;
#pragma unroll
  for (int off = 16; off > 0; off >>= 1) ss += __shfl_xor(ss, off, 32);
  float kmag = sqrtf(ss);

  size_t fbase = ((size_t)bh * Nq + n) * HDq;
  field[fbase + lane]      = vp[lane] * kmag;
  field[fbase + lane + 32] = vp[lane + 32] * kmag;

  float q0 = qp[lane], q1 = qp[lane + 32];
  float logit[Sq];
#pragma unroll
  for (int s = 0; s < Sq; ++s) {
    float p = q0 * Wqs[s * HDq + lane] + q1 * Wqs[s * HDq + lane + 32];
#pragma unroll
    for (int off = 16; off > 0; off >>= 1) p += __shfl_xor(p, off, 32);
    logit[s] = p + scale_gain[s * Hq + h];
  }
  float mx = logit[0];
#pragma unroll
  for (int s = 1; s < Sq; ++s) mx = fmaxf(mx, logit[s]);
  float sum = 0.f;
#pragma unroll
  for (int s = 0; s < Sq; ++s) { logit[s] = __expf(logit[s] - mx); sum += logit[s]; }
  float inv = 1.0f / sum;
  if (lane < Sq) gains[((size_t)bh * Nq + n) * Sq + lane] = logit[lane] * inv;
}

// ---------------------------------------------------------------------------
// Causal multiscale dilated D4 conv, gain-weighted accumulate.
// ---------------------------------------------------------------------------
__launch_bounds__(256)
__global__ void wavelet_kernel(const float* __restrict__ field,  // [B,H,N,HD]
                               const float* __restrict__ gains,  // [B,H,N,S]
                               float* __restrict__ outf) {       // [B,H,N,HD]
  int bh    = blockIdx.x / (Nq / 4);
  int ntile = blockIdx.x % (Nq / 4);
  int n     = ntile * 4 + (threadIdx.x >> 6);
  int hd    = threadIdx.x & 63;
  const float* f = field + (size_t)bh * Nq * HDq;
  const float* g = gains + ((size_t)bh * Nq + n) * Sq;
  float acc = 0.f;
  int d = 1;
#pragma unroll
  for (int j = 0; j < Sq; ++j) {
    int n3 = n - 3 * d, n2 = n - 2 * d, n1 = n - d;
    float y = -0.1294095225512604f * f[(size_t)n * HDq + hd];
    if (n1 >= 0) y += 0.2241438680420134f * f[(size_t)n1 * HDq + hd];
    if (n2 >= 0) y += 0.8365163037378079f * f[(size_t)n2 * HDq + hd];
    if (n3 >= 0) y += 0.4829629131445341f * f[(size_t)n3 * HDq + hd];
    acc += g[j] * y;
    d <<= 1;
  }
  outf[((size_t)bh * Nq + n) * HDq + hd] = acc;
}

// ---------------------------------------------------------------------------
// 16x16 row softmax of field_coupling (tiny, one wave)
// ---------------------------------------------------------------------------
__global__ void coupling_softmax_kernel(const float* __restrict__ fc,
                                        float* __restrict__ cp) {
  int i = threadIdx.x;
  if (i < Hq) {
    float mx = -1e30f;
    for (int j = 0; j < Hq; ++j) mx = fmaxf(mx, fc[i * Hq + j]);
    float e[Hq];
    float sum = 0.f;
    for (int j = 0; j < Hq; ++j) { e[j] = __expf(fc[i * Hq + j] - mx); sum += e[j]; }
    float inv = 1.0f / sum;
    for (int j = 0; j < Hq; ++j) cp[i * Hq + j] = e[j] * inv;
  }
}

// ---------------------------------------------------------------------------
// Head mixing (coupling) + sigmoid gate multiply -> bf16 A operand [B*N, D].
// ---------------------------------------------------------------------------
__launch_bounds__(256)
__global__ void mix_gate_kernel(const float* __restrict__ outf,  // [B,H,N,HD]
                                const float* __restrict__ coup,  // [H,H]
                                const float* __restrict__ gate,  // [B*N, D]
                                __bf16* __restrict__ gg) {       // [B*N, D]
  __shared__ float cs[Hq * Hq];
  if (threadIdx.x < Hq * Hq) cs[threadIdx.x] = coup[threadIdx.x];
  __syncthreads();
  int bn = blockIdx.x;
  int b = bn / Nq, n = bn % Nq;
#pragma unroll
  for (int rep = 0; rep < 4; ++rep) {
    int d  = threadIdx.x + rep * 256;
    int i  = d >> 6, hd = d & 63;
    float s = 0.f;
#pragma unroll
    for (int j = 0; j < Hq; ++j)
      s += cs[i * Hq + j] * outf[(((size_t)b * Hq + j) * Nq + n) * HDq + hd];
    size_t o = (size_t)bn * Dq + d;
    gg[o] = (__bf16)(s * gate[o]);
  }
}

// ---------------------------------------------------------------------------
// Host-side orchestration
// ---------------------------------------------------------------------------
extern "C" void kernel_launch(void* const* d_in, const int* in_sizes, int n_in,
                              void* d_out, int out_size, void* d_ws, size_t ws_size,
                              hipStream_t stream) {
  const float* x          = (const float*)d_in[0];
  const float* Wqkv       = (const float*)d_in[1];
  const float* bqkv       = (const float*)d_in[2];
  const float* Wo         = (const float*)d_in[3];
  const float* bo         = (const float*)d_in[4];
  const float* Wg         = (const float*)d_in[5];
  const float* bg         = (const float*)d_in[6];
  const float* scale_gain = (const float*)d_in[7];
  const float* Wqs        = (const float*)d_in[8];
  const float* fcoup      = (const float*)d_in[9];
  float* out = (float*)d_out;

  char* w = (char*)d_ws;
  auto carve = [&](size_t bytes) {
    char* p = w;
    w += (bytes + 255) & ~(size_t)255;
    return p;
  };
  __bf16* xbf    = (__bf16*)carve((size_t)Mq * Kq * 2);
  __bf16* wqkvbf = (__bf16*)carve((size_t)3 * Dq * Dq * 2);
  __bf16* wgbf   = (__bf16*)carve((size_t)Dq * Dq * 2);
  __bf16* wobf   = (__bf16*)carve((size_t)Dq * Dq * 2);
  float*  qkv    = (float*)carve((size_t)Mq * 3 * Dq * 4);
  float*  gate   = (float*)carve((size_t)Mq * Dq * 4);
  float*  gains  = (float*)carve((size_t)Bq * Hq * Nq * Sq * 4);
  float*  field  = (float*)carve((size_t)Bq * Hq * Nq * HDq * 4);
  float*  outf   = (float*)carve((size_t)Bq * Hq * Nq * HDq * 4);
  float*  coup   = (float*)carve((size_t)Hq * Hq * 4);
  __bf16* gg     = (__bf16*)carve((size_t)Mq * Dq * 2);

  auto cvtLaunch = [&](const float* src, __bf16* dst, int n) {
    cvt_bf16_kernel<<<(n / 4 + 255) / 256, 256, 0, stream>>>(src, dst, n);
  };
  cvtLaunch(x, xbf, Mq * Kq);
  cvtLaunch(Wqkv, wqkvbf, 3 * Dq * Dq);
  cvtLaunch(Wg, wgbf, Dq * Dq);
  cvtLaunch(Wo, wobf, Dq * Dq);

  coupling_softmax_kernel<<<1, 32, 0, stream>>>(fcoup, coup);

  dim3 g1(Mq / BM, (3 * Dq) / BN);
  gemm_bf16_wmma_kernel<<<g1, 256, 0, stream>>>(xbf, wqkvbf, bqkv, qkv,
                                                Mq, 3 * Dq, Kq, 0);
  dim3 g2(Mq / BM, Dq / BN);
  gemm_bf16_wmma_kernel<<<g2, 256, 0, stream>>>(xbf, wgbf, bg, gate,
                                                Mq, Dq, Kq, 1);

  gains_field_kernel<<<(Bq * Hq * Nq) / 8, 256, 0, stream>>>(qkv, scale_gain,
                                                             Wqs, gains, field);
  wavelet_kernel<<<Bq * Hq * (Nq / 4), 256, 0, stream>>>(field, gains, outf);
  mix_gate_kernel<<<Bq * Nq, 256, 0, stream>>>(outf, coup, gate, gg);

  dim3 g3(Mq / BM, Dq / BN);
  gemm_bf16_wmma_kernel<<<g3, 256, 0, stream>>>(gg, wobf, bo, out,
                                                Mq, Dq, Kq, 0);

  (void)in_sizes; (void)n_in; (void)out_size; (void)ws_size;
}
